// XNORConv2d_1219770712410
// MI455X (gfx1250) — 8-bit matrix operand order repaired and checked by reading, compiled, not run on hardware
//
#include <hip/hip_runtime.h>

#define B_   32
#define CIN  256
#define COUT 256
#define H_   56
#define W_   56
#define HW   3136
#define HP   58
#define WP   58
#define KRED 2304
#define NKC  36

#define MT   64
#define NT   128
#define AST  80
#define A_BYTES   (64 * AST)
#define B_BYTES   (128 * AST)
#define BUF_BYTES (A_BYTES + B_BYTES)
#define NBUF 3

typedef __attribute__((ext_vector_type(8))) int v8i;
typedef __attribute__((ext_vector_type(4))) int v4i_t;

#define XPAD_BYTES (32u * 58u * 58u * 256u)
#define WT_BYTES   (256u * 2304u)

#ifndef __has_builtin
#define __has_builtin(x) 0
#endif
#if __has_builtin(__builtin_amdgcn_global_load_async_to_lds_b128) && \
    __has_builtin(__builtin_amdgcn_s_wait_asynccnt)
#define HAVE_ASYNC 1
#else
#define HAVE_ASYNC 0
#endif

__global__ void zero_ws(int4* __restrict__ p, int n16) {
    int i = blockIdx.x * blockDim.x + threadIdx.x;
    int stride = gridDim.x * blockDim.x;
    int4 z = make_int4(0, 0, 0, 0);
    for (; i < n16; i += stride) p[i] = z;
}

__global__ void prep_w(const float* __restrict__ w,
                       signed char* __restrict__ wt,
                       float* __restrict__ alpha) {
    const int co = blockIdx.x;
    const int ci = threadIdx.x;
    __shared__ float red[256];
    float aacc = 0.f;
    for (int kk = 0; kk < 9; ++kk) {
        float v = w[(size_t)(co * 256 + ci) * 9 + kk];
        red[ci] = v;
        __syncthreads();
        for (int off = 128; off > 0; off >>= 1) {
            if (ci < off) red[ci] += red[ci + off];
            __syncthreads();
        }
        float mean = red[0] * (1.0f / 256.0f);
        __syncthreads();
        float wc = v - mean;
        signed char s = (wc > 0.f) ? 1 : ((wc < 0.f) ? -1 : 0);
        wt[(size_t)co * KRED + kk * 256 + ci] = s;
        aacc += fminf(fabsf(wc), 1.0f);
    }
    red[ci] = aacc;
    __syncthreads();
    for (int off = 128; off > 0; off >>= 1) {
        if (ci < off) red[ci] += red[ci + off];
        __syncthreads();
    }
    if (ci == 0) alpha[co] = red[0] * (1.0f / 2304.0f);
}

__global__ void binarize_x(const float* __restrict__ x,
                           signed char* __restrict__ xpad) {
    const int b  = blockIdx.z;
    const int c0 = blockIdx.y * 64;
    const int p0 = blockIdx.x * 64;
    __shared__ signed char s[64][65];
    const int t = threadIdx.x;
    for (int it = 0; it < 16; ++it) {
        int e  = it * 256 + t;
        int cl = e >> 6;
        int pl = e & 63;
        float v = x[(size_t)(b * 256 + c0 + cl) * HW + p0 + pl];
        s[cl][pl] = (v > 0.f) ? 1 : ((v < 0.f) ? -1 : 0);
    }
    __syncthreads();
    for (int it = 0; it < 16; ++it) {
        int e  = it * 256 + t;
        int pl = e >> 6;
        int cl = e & 63;
        int P  = p0 + pl;
        int y  = P / 56;
        int xc = P - y * 56;
        xpad[(size_t)((b * HP + (y + 1)) * WP + (xc + 1)) * 256 + c0 + cl] = s[cl][pl];
    }
}

__global__ __launch_bounds__(256)
void xnor_conv(const signed char* __restrict__ xpad,
               const signed char* __restrict__ wt,
               const float* __restrict__ alpha,
               const float* __restrict__ bias,
               float* __restrict__ out) {
    __shared__ __align__(16) signed char smem[NBUF * BUF_BYTES];

    const int t      = threadIdx.x;
    const int m_base = blockIdx.x * MT;
    const int n_base = blockIdx.y * NT;

    const int lrow = t >> 2;
    const int lseg = (t & 3) * 16;
    int P    = m_base + lrow;
    int pb   = P / HW;
    int prem = P - pb * HW;
    int py   = prem / 56;
    int px   = prem - py * 56;
    const signed char* apix  = xpad + ((size_t)(pb * HP + py) * WP + px) * 256 + lseg;
    const signed char* brow0 = wt + (size_t)(n_base + lrow) * KRED + lseg;
    const signed char* brow1 = brow0 + (size_t)64 * KRED;
    const int sAoff  = lrow * AST + lseg;
    const int sBoff0 = A_BYTES + lrow * AST + lseg;
    const int sBoff1 = sBoff0 + 64 * AST;

    const int lane = t & 31;
    const int wv   = t >> 5;
    const int wm   = wv & 3;
    const int wn   = wv >> 2;
    const int lx   = lane & 15;
    const int lh   = lane >> 4;
    const int aFragOff = (wm * 16 + lx) * AST + lh * 16;
    const int bFragBase = A_BYTES + (wn * 64 + lx) * AST + lh * 16;

    v8i acc0 = (v8i){0, 0, 0, 0, 0, 0, 0, 0};
    v8i acc1 = acc0, acc2 = acc0, acc3 = acc0;

    auto a_off = [](int kc) {
        int khkw = kc >> 2;
        int kh   = khkw / 3;
        int kw   = khkw - kh * 3;
        return (kh * WP + kw) * 256 + (kc & 3) * 64;
    };

    auto compute = [&](const signed char* bufp) {
        union { unsigned long long q[4]; v8i v; } ua;
        const signed char* ap = bufp + aFragOff;
        ua.q[0] = *(const unsigned long long*)(ap + 0);
        ua.q[1] = *(const unsigned long long*)(ap + 8);
        ua.q[2] = *(const unsigned long long*)(ap + 32);
        ua.q[3] = *(const unsigned long long*)(ap + 40);
        union { int4 h[2]; v8i v; } ub0, ub1, ub2, ub3;
        const signed char* bp = bufp + bFragBase;
        ub0.h[0] = *(const int4*)(bp + 0 * 16 * AST);
        ub0.h[1] = *(const int4*)(bp + 0 * 16 * AST + 32);
        ub1.h[0] = *(const int4*)(bp + 1 * 16 * AST);
        ub1.h[1] = *(const int4*)(bp + 1 * 16 * AST + 32);
        ub2.h[0] = *(const int4*)(bp + 2 * 16 * AST);
        ub2.h[1] = *(const int4*)(bp + 2 * 16 * AST + 32);
        ub3.h[0] = *(const int4*)(bp + 3 * 16 * AST);
        ub3.h[1] = *(const int4*)(bp + 3 * 16 * AST + 32);
        acc0 = __builtin_amdgcn_wmma_i32_16x16x64_iu8(true, ua.v, true, ub0.v,
                                                      acc0, false, false);
        acc1 = __builtin_amdgcn_wmma_i32_16x16x64_iu8(true, ua.v, true, ub1.v,
                                                      acc1, false, false);
        acc2 = __builtin_amdgcn_wmma_i32_16x16x64_iu8(true, ua.v, true, ub2.v,
                                                      acc2, false, false);
        acc3 = __builtin_amdgcn_wmma_i32_16x16x64_iu8(true, ua.v, true, ub3.v,
                                                      acc3, false, false);
    };

#if HAVE_ASYNC
    auto issue = [&](int kc, int buf) {
        signed char* sbuf = smem + buf * BUF_BYTES;
        int ao = a_off(kc);
        int bo = kc * 64;
        __builtin_amdgcn_global_load_async_to_lds_b128(
            (v4i_t*)(apix + ao),  (v4i_t*)(sbuf + sAoff), 0, 0);
        __builtin_amdgcn_global_load_async_to_lds_b128(
            (v4i_t*)(brow0 + bo), (v4i_t*)(sbuf + sBoff0), 0, 0);
        __builtin_amdgcn_global_load_async_to_lds_b128(
            (v4i_t*)(brow1 + bo), (v4i_t*)(sbuf + sBoff1), 0, 0);
    };

    issue(0, 0);
    int cur = 0;
    for (int kc = 0; kc < NKC; ++kc) {
        int nxt = cur + 1;
        if (nxt == NBUF) nxt = 0;
        if (kc + 1 < NKC) {
            issue(kc + 1, nxt);
            __builtin_amdgcn_s_wait_asynccnt(3);
        } else {
            __builtin_amdgcn_s_wait_asynccnt(0);
        }
        __syncthreads();
        compute(smem + cur * BUF_BYTES);
        cur = nxt;
    }
#else
    for (int kc = 0; kc < NKC; ++kc) {
        int ao = a_off(kc);
        int bo = kc * 64;
        int4 ra  = *(const int4*)(apix + ao);
        int4 rb0 = *(const int4*)(brow0 + bo);
        int4 rb1 = *(const int4*)(brow1 + bo);
        __syncthreads();
        *(int4*)(smem + sAoff)  = ra;
        *(int4*)(smem + sBoff0) = rb0;
        *(int4*)(smem + sBoff1) = rb1;
        __syncthreads();
        compute(smem);
    }
#endif

    const int n_g   = n_base + wn * 64 + lx;
    const int Pbase = m_base + wm * 16 + lh * 8;
    v8i accs[4] = {acc0, acc1, acc2, acc3};
#pragma unroll
    for (int s = 0; s < 4; ++s) {
        int n = n_g + s * 16;
        float sc = alpha[n], bi = bias[n];
#pragma unroll
        for (int r = 0; r < 8; ++r) {
            int Pm = Pbase + r;
            int bb = Pm / HW;
            int rr = Pm - bb * HW;
            out[(size_t)(bb * COUT + n) * HW + rr] = sc * (float)accs[s][r] + bi;
        }
    }
}

extern "C" void kernel_launch(void* const* d_in, const int* in_sizes, int n_in,
                              void* d_out, int out_size, void* d_ws, size_t ws_size,
                              hipStream_t stream) {
    const float* x    = (const float*)d_in[0];
    const float* w    = (const float*)d_in[1];
    const float* bias = (const float*)d_in[2];
    float* out        = (float*)d_out;

    char* ws           = (char*)d_ws;
    signed char* xpad  = (signed char*)ws;
    signed char* wtb   = (signed char*)(ws + XPAD_BYTES);
    float*       alpha = (float*)(ws + XPAD_BYTES + WT_BYTES);

    zero_ws<<<512, 256, 0, stream>>>((int4*)xpad, (int)(XPAD_BYTES / 16));
    prep_w<<<COUT, 256, 0, stream>>>(w, wtb, alpha);
    binarize_x<<<dim3(49, 4, 32), 256, 0, stream>>>(x, xpad);
    xnor_conv<<<dim3(1568, 2), 256, 0, stream>>>(xpad, wtb, alpha, bias, out);
}
